// HLGAttention_83897891160258
// MI455X (gfx1250) — matrix operand order repaired and checked by reading, not compiled, not run on hardware
//
#include <hip/hip_runtime.h>

// ---------------- problem constants ----------------
#define GS 14
#define NPOS 196                 // GS*GS
#define C_DIM 128
#define H_HEADS 4
#define HD 32
#define BATCH 1024
#define MTOT (BATCH * NPOS)      // 200704 = 12544 * 16 exactly
#define MTILES (MTOT / 16)       // 12544
#define QK_LD 256                // Q|K concatenated columns (row major)
#define NKT 13                   // ceil(196/16) key tiles
#define KT_PAD 224               // 7*32 padded key length
#define POS_TAB 729              // (2*GS-1)^2
#define SCALE 0.1767766953f      // 1/sqrt(32)
#define NEG_BIG (-1.0e38f)

typedef __attribute__((ext_vector_type(16))) __bf16 v16bf;
typedef __attribute__((ext_vector_type(8)))  float  v8f;
typedef __attribute__((ext_vector_type(4)))  int    v4i;

struct U32x8 { uint4 a, b; };

static __device__ __forceinline__ v16bf v16bf_from(uint4 x, uint4 y) {
  U32x8 t; t.a = x; t.b = y;
  return __builtin_bit_cast(v16bf, t);
}

static __device__ __forceinline__ unsigned short f2bfu(float f) {
  unsigned u = __builtin_bit_cast(unsigned, f);
  unsigned r = u + 0x7FFFu + ((u >> 16) & 1u);      // round-to-nearest-even
  return (unsigned short)(r >> 16);
}
static __device__ __forceinline__ __bf16 f2bf(float f) {
  unsigned short h = f2bfu(f);
  return __builtin_bit_cast(__bf16, h);
}
// packed f32x2 -> bf16x2 using CDNA5 VOP3P V_FMA_MIXLO/MIXHI_BF16 (ISA 15.10)
static __device__ __forceinline__ unsigned cvt2(float lo, float hi) {
  unsigned r = 0;
  asm("v_fma_mixlo_bf16 %0, %1, 1.0, 0" : "+v"(r) : "v"(lo));
  asm("v_fma_mixhi_bf16 %0, %1, 1.0, 0" : "+v"(r) : "v"(hi));
  return r;
}

// -------- async global->LDS staging (CDNA5) --------
#if __has_builtin(__builtin_amdgcn_global_load_async_to_lds_b128) && \
    __has_builtin(__builtin_amdgcn_s_wait_asynccnt)
#define HAS_ASYNC_LDS 1
#else
#define HAS_ASYNC_LDS 0
#endif

static __device__ __forceinline__ void stage_b128(const uint4* gsrc, uint4* ldst) {
#if HAS_ASYNC_LDS
  __builtin_amdgcn_global_load_async_to_lds_b128(
      (__attribute__((address_space(1))) v4i*)(v4i*)(uint4*)gsrc,
      (__attribute__((address_space(3))) v4i*)(v4i*)ldst, 0, 0);
#else
  *ldst = *gsrc;
#endif
}
static __device__ __forceinline__ void stage_wait() {
#if HAS_ASYNC_LDS
  __builtin_amdgcn_s_wait_asynccnt(0);
#endif
  __syncthreads();
}

// ---------------- kernel: zero fill (uint4 granularity) ----------------
__global__ void hlg_fill0(uint4* __restrict__ p, int n16) {
  int i = blockIdx.x * blockDim.x + threadIdx.x;
  if (i < n16) { uint4 z = {0u, 0u, 0u, 0u}; p[i] = z; }
}

// ---------------- kernel: weights -> bf16 TRANSPOSED ----------------
// wqkvT[col][k] for col in [0,384), wprojT[col][k] for col in [0,128)
__global__ void hlg_cvt_w(const float* __restrict__ wq, const float* __restrict__ wkv,
                          const float* __restrict__ wproj,
                          __bf16* __restrict__ wqkvT, __bf16* __restrict__ wprojT) {
  int idx = blockIdx.x * blockDim.x + threadIdx.x;    // < 65536
  if (idx < 384 * C_DIM) {
    int col = idx >> 7, k = idx & 127;
    float w = (col < C_DIM) ? wq[(size_t)k * C_DIM + col]
                            : wkv[(size_t)k * 2 * C_DIM + (col - C_DIM)];
    wqkvT[(size_t)col * C_DIM + k] = f2bf(w);
  } else {
    int i2 = idx - 384 * C_DIM;                       // < 16384
    int col = i2 >> 7, k = i2 & 127;
    wprojT[(size_t)col * C_DIM + k] = f2bf(wproj[(size_t)k * C_DIM + col]);
  }
}

// ---------------- kernel: dynamic position-bias MLP ----------------
static __device__ __forceinline__ void ln_relu8(const float* x, const float* g,
                                                const float* b, float* y) {
  float mu = 0.f;
#pragma unroll
  for (int i = 0; i < 8; ++i) mu += x[i];
  mu *= 0.125f;
  float v = 0.f;
#pragma unroll
  for (int i = 0; i < 8; ++i) { float d = x[i] - mu; v += d * d; }
  v *= 0.125f;
  float r = rsqrtf(v + 1e-5f);
#pragma unroll
  for (int i = 0; i < 8; ++i) {
    float f = (x[i] - mu) * r * g[i] + b[i];
    y[i] = f > 0.f ? f : 0.f;
  }
}

__global__ void hlg_pos_mlp(const float* __restrict__ pp_w, const float* __restrict__ pp_b,
                            const float* __restrict__ g1, const float* __restrict__ b1,
                            const float* __restrict__ w1, const float* __restrict__ bb1,
                            const float* __restrict__ g2, const float* __restrict__ b2,
                            const float* __restrict__ w2, const float* __restrict__ bb2,
                            const float* __restrict__ g3, const float* __restrict__ b3,
                            const float* __restrict__ w3, const float* __restrict__ bb3,
                            float* __restrict__ rpb) {
  int t = blockIdx.x * blockDim.x + threadIdx.x;
  if (t >= POS_TAB) return;
  float c0 = (float)(t / 27 - 13);
  float c1 = (float)(t % 27 - 13);
  float h[8], y[8];
#pragma unroll
  for (int j = 0; j < 8; ++j) h[j] = c0 * pp_w[j] + c1 * pp_w[8 + j] + pp_b[j];
  ln_relu8(h, g1, b1, y);
#pragma unroll
  for (int j = 0; j < 8; ++j) {
    float s = bb1[j];
#pragma unroll
    for (int i = 0; i < 8; ++i) s += y[i] * w1[i * 8 + j];
    h[j] = s;
  }
  ln_relu8(h, g2, b2, y);
#pragma unroll
  for (int j = 0; j < 8; ++j) {
    float s = bb2[j];
#pragma unroll
    for (int i = 0; i < 8; ++i) s += y[i] * w2[i * 8 + j];
    h[j] = s;
  }
  ln_relu8(h, g3, b3, y);
#pragma unroll
  for (int hh = 0; hh < H_HEADS; ++hh) {
    float s = bb3[hh];
#pragma unroll
    for (int i = 0; i < 8; ++i) s += y[i] * w3[i * H_HEADS + hh];
    rpb[t * H_HEADS + hh] = s;
  }
}

// ------- kernel: expand bias, TILED layout Bt[h][qtile][kcol][m(16)] -------
__global__ void hlg_bias_expand(const float* __restrict__ rpb, float* __restrict__ Bt) {
  int idx = blockIdx.x * blockDim.x + threadIdx.x;
  if (idx >= H_HEADS * NKT * KT_PAD * 16) return;
  int m  = idx & 15;
  int t  = idx >> 4;
  int kc = t % KT_PAD;
  int t2 = t / KT_PAD;
  int qt = t2 % NKT;
  int h  = t2 / NKT;
  int q  = qt * 16 + m;
  float v;
  if (q >= NPOS)       v = 0.f;
  else if (kc >= NPOS) v = NEG_BIG;                   // masks padded keys
  else {
    int qi = q / GS, qj = q - qi * GS;
    int ki = kc / GS, kj = kc - ki * GS;
    v = rpb[((qi - ki + GS - 1) * (2 * GS - 1) + (qj - kj + GS - 1)) * H_HEADS + h];
  }
  Bt[idx] = v;
}

// ---------------- kernel: fused QKV projection (bf16 WMMA) ----------------
// x f32 converted during LDS staging; one block covers all 24 column tiles.
__global__ void hlg_qkv_gemm(const float* __restrict__ x,
                             const __bf16* __restrict__ wqkvT,
                             __bf16* __restrict__ qk,
                             __bf16* __restrict__ vt) {
  __shared__ __bf16 xt[16][C_DIM];                    // 4 KB bf16 X tile
  const int tid  = threadIdx.x;
  const int wave = tid >> 5;
  const int lane = tid & 31;
  const int l15  = lane & 15;
  const int hi   = lane >> 4;
  const int mbase = blockIdx.x * 16;

  // fused f32->bf16 staging: each thread converts 8 elements
  {
    const float4* xf = (const float4*)(x + (size_t)mbase * C_DIM) + 2 * tid;
    float4 f0 = xf[0], f1 = xf[1];
    uint4 o;
    o.x = cvt2(f0.x, f0.y); o.y = cvt2(f0.z, f0.w);
    o.z = cvt2(f1.x, f1.y); o.w = cvt2(f1.z, f1.w);
    ((uint4*)xt)[tid] = o;
  }
  __syncthreads();

  // hoist A fragments (reused across all 3 column groups)
  v16bf afr[4];
#pragma unroll
  for (int kc = 0; kc < 4; ++kc) {
    const uint4* ar = (const uint4*)&xt[l15][kc * 32 + hi * 16];
    afr[kc] = v16bf_from(ar[0], ar[1]);
  }

  // scalar (block-uniform) batch/pos decomposition for the V path
  const int sb  = mbase / NPOS;
  const int snn = mbase - sb * NPOS;

#pragma unroll
  for (int g = 0; g < 3; ++g) {                       // column groups: Q, K, V
    const int colbase = (wave + 8 * g) * 16;
    v8f acc = {};
#pragma unroll
    for (int kc = 0; kc < 4; ++kc) {
      const uint4* br = (const uint4*)(wqkvT + (size_t)(colbase + l15) * C_DIM + kc * 32 + hi * 16);
      v16bf bt = v16bf_from(br[0], br[1]);
      acc = __builtin_amdgcn_wmma_f32_16x16x32_bf16(false, afr[kc], false, bt,
                                                    (short)0, acc, false, false);
    }
    if (g < 2) {                                      // Q|K rows (compile-time)
      __bf16* p = qk + (size_t)(mbase + hi * 8) * QK_LD + colbase + l15;
#pragma unroll
      for (int r = 0; r < 8; r += 2) {
        unsigned pk = cvt2(acc[r], acc[r + 1]);
        *(unsigned short*)(p + (size_t)r * QK_LD)       = (unsigned short)pk;
        *(unsigned short*)(p + (size_t)(r + 1) * QK_LD) = (unsigned short)(pk >> 16);
      }
    } else {                                          // V transposed (compile-time)
      const int cv = colbase + l15 - 2 * C_DIM;       // head*32 + d
      __bf16* vl = vt + (size_t)cv * KT_PAD;
#pragma unroll
      for (int r = 0; r < 8; r += 2) {
        unsigned pk = cvt2(acc[r], acc[r + 1]);
        int m0 = hi * 8 + r;
        int n0 = snn + m0;     int b0 = sb + (n0 >= NPOS); n0 -= (n0 >= NPOS) ? NPOS : 0;
        int n1 = snn + m0 + 1; int b1 = sb + (n1 >= NPOS); n1 -= (n1 >= NPOS) ? NPOS : 0;
        *(unsigned short*)(vl + (size_t)b0 * (C_DIM * KT_PAD) + n0) = (unsigned short)pk;
        *(unsigned short*)(vl + (size_t)b1 * (C_DIM * KT_PAD) + n1) = (unsigned short)(pk >> 16);
      }
    }
  }
}

// ---------------- kernel: attention core ----------------
__global__ void __launch_bounds__(32)
hlg_attn(const __bf16* __restrict__ qk,
         const __bf16* __restrict__ vt,
         const float* __restrict__ Bt,
         __bf16* __restrict__ aout) {
  __shared__ float    Sf[KT_PAD][16];                 // 14 KB scores (key-major!)
  __shared__ unsigned Pbu[16][KT_PAD / 2];            // 7 KB packed bf16 probs

  const int lane = threadIdx.x;
  const int l15  = lane & 15;
  const int hi   = lane >> 4;
  const int qt = blockIdx.x;                          // 0..12
  const int h  = blockIdx.y;                          // 0..3
  const int b  = blockIdx.z;                          // 0..1023
  const int qbase = qt * 16;
  const size_t rowb = (size_t)b * NPOS;
  // tiled bias: Bt[h][qt][kcol][m]; this lane's 8 values are 32 contig bytes
  const float* btq = Bt + ((size_t)(h * NKT + qt) * KT_PAD) * 16;

  // ---- Q fragment (A layout; K-dim = hd = 32 -> single WMMA step) ----
  v16bf aq;
  {
    const int qg = qbase + l15;
    const __bf16* qrow = qk + (rowb + (qg < NPOS ? qg : NPOS - 1)) * QK_LD + h * HD;
    const uint4* qr = (const uint4*)(qrow + hi * 16);
    aq = v16bf_from(qr[0], qr[1]);
  }

  // ---- S = Q K^T * scale + bias -> LDS (two b128 stores per tile) ----
  for (int kt = 0; kt < NKT; ++kt) {
    const int kbase = kt * 16;
    const int kgc = (kbase + l15 < NPOS) ? kbase + l15 : NPOS - 1;
    if (kt + 1 < NKT)
      __builtin_prefetch(qk + (rowb + ((kbase + 16 + l15 < NPOS) ? kbase + 16 + l15 : NPOS - 1)) * QK_LD + C_DIM + h * HD, 0, 1);
    const uint4* kr = (const uint4*)(qk + (rowb + kgc) * QK_LD + C_DIM + h * HD + hi * 16);
    v16bf bk = v16bf_from(kr[0], kr[1]);

    v8f zero = {};
    v8f s = __builtin_amdgcn_wmma_f32_16x16x32_bf16(false, aq, false, bk,
                                                    (short)0, zero, false, false);
    const float4* bp = (const float4*)(btq + (size_t)(kbase + l15) * 16 + hi * 8);
    float4 bA = bp[0], bB = bp[1];
    float4 s0 = {fmaf(s[0], SCALE, bA.x), fmaf(s[1], SCALE, bA.y),
                 fmaf(s[2], SCALE, bA.z), fmaf(s[3], SCALE, bA.w)};
    float4 s1 = {fmaf(s[4], SCALE, bB.x), fmaf(s[5], SCALE, bB.y),
                 fmaf(s[6], SCALE, bB.z), fmaf(s[7], SCALE, bB.w)};
    float4* sp = (float4*)&Sf[kbase + l15][hi * 8];
    sp[0] = s0; sp[1] = s1;
  }
#pragma unroll
  for (int i = lane; i < 16 * 16; i += 32)            // pad keys 208..223
    Sf[(NKT * 16) + (i >> 4)][i & 15] = NEG_BIG;
  __syncthreads();

  // ---- softmax: lane pair (L, L^16) shares row L&15, 112 keys each ----
  {
    const int row = l15;
    const int c0 = hi * (KT_PAD / 2);
    float pmax = NEG_BIG;
    for (int c = 0; c < KT_PAD / 2; ++c) pmax = fmaxf(pmax, Sf[c0 + c][row]);
    float tmax = fmaxf(pmax, __shfl_xor(pmax, 16));
    float psum = 0.f;
    for (int c = 0; c < KT_PAD / 2; ++c) {
      float e = __expf(Sf[c0 + c][row] - tmax);
      Sf[c0 + c][row] = e;
      psum += e;
    }
    float tsum = psum + __shfl_xor(psum, 16);
    float rinv = 1.0f / tsum;
    for (int c = 0; c < KT_PAD / 2; c += 2)
      Pbu[row][(c0 + c) >> 1] = cvt2(Sf[c0 + c][row] * rinv, Sf[c0 + c + 1][row] * rinv);
  }
  __syncthreads();

  // ---- O = P @ V : 7 K-chunks of 32 keys, 2 output tiles (hd=32) ----
  v8f acc0 = {}, acc1 = {};
  const __bf16* vbase = vt + (((size_t)b * H_HEADS + h) * HD) * KT_PAD;
  for (int kc = 0; kc < KT_PAD / 32; ++kc) {
    const int kb = kc * 32;
    const unsigned* pr = &Pbu[l15][(kb >> 1) + hi * 8];
    v16bf ap = v16bf_from(*(const uint4*)pr, *(const uint4*)(pr + 4));
    const uint4* v0 = (const uint4*)(vbase + (size_t)l15 * KT_PAD + kb + hi * 16);
    const uint4* v1 = (const uint4*)(vbase + (size_t)(16 + l15) * KT_PAD + kb + hi * 16);
    v16bf bv0 = v16bf_from(v0[0], v0[1]);
    v16bf bv1 = v16bf_from(v1[0], v1[1]);
    acc0 = __builtin_amdgcn_wmma_f32_16x16x32_bf16(false, ap, false, bv0,
                                                   (short)0, acc0, false, false);
    acc1 = __builtin_amdgcn_wmma_f32_16x16x32_bf16(false, ap, false, bv1,
                                                   (short)0, acc1, false, false);
  }

#pragma unroll
  for (int r = 0; r < 8; r += 2) {
    unsigned p0 = cvt2(acc0[r], acc0[r + 1]);
    unsigned p1 = cvt2(acc1[r], acc1[r + 1]);
    const int qg0 = qbase + r + hi * 8;
    const int qg1 = qg0 + 1;
    __bf16* row0 = aout + (rowb + qg0) * C_DIM + h * HD;
    if (qg0 < NPOS) {
      *(unsigned short*)(row0 + l15)      = (unsigned short)p0;
      *(unsigned short*)(row0 + 16 + l15) = (unsigned short)p1;
    }
    if (qg1 < NPOS) {
      *(unsigned short*)(row0 + C_DIM + l15)      = (unsigned short)(p0 >> 16);
      *(unsigned short*)(row0 + C_DIM + 16 + l15) = (unsigned short)(p1 >> 16);
    }
  }
}

// ---------------- kernel: output projection ----------------
__global__ void hlg_proj_gemm(const __bf16* __restrict__ a_in,
                              const __bf16* __restrict__ wprojT,
                              const float* __restrict__ bproj,
                              float* __restrict__ out) {
  __shared__ __bf16 at[16][C_DIM];                    // 4 KB A tile
  const int tid  = threadIdx.x;
  const int wave = tid >> 5;
  const int lane = tid & 31;
  const int l15  = lane & 15;
  const int hi   = lane >> 4;
  const int mbase = blockIdx.x * 16;
  const int colbase = wave * 16;                      // 8 waves -> 128 cols

  stage_b128((const uint4*)(a_in + (size_t)mbase * C_DIM) + tid, ((uint4*)at) + tid);
  stage_wait();

  v8f acc = {};
#pragma unroll
  for (int kc = 0; kc < 4; ++kc) {
    const int k0 = kc * 32;
    const uint4* ar = (const uint4*)&at[l15][k0 + hi * 16];
    v16bf a = v16bf_from(ar[0], ar[1]);
    const uint4* br = (const uint4*)(wprojT + (size_t)(colbase + l15) * C_DIM + k0 + hi * 16);
    v16bf bt = v16bf_from(br[0], br[1]);
    acc = __builtin_amdgcn_wmma_f32_16x16x32_bf16(false, a, false, bt,
                                                  (short)0, acc, false, false);
  }
  const float bias = bproj[colbase + l15];
  float* p = out + (size_t)(mbase + hi * 8) * C_DIM + colbase + l15;
#pragma unroll
  for (int r = 0; r < 8; ++r)
    p[(size_t)r * C_DIM] = acc[r] + bias;
}

// ---------------- host launcher ----------------
extern "C" void kernel_launch(void* const* d_in, const int* in_sizes, int n_in,
                              void* d_out, int out_size, void* d_ws, size_t ws_size,
                              hipStream_t stream) {
  const float* x     = (const float*)d_in[0];
  const float* wq    = (const float*)d_in[1];
  const float* wkv   = (const float*)d_in[2];
  const float* wproj = (const float*)d_in[3];
  const float* bproj = (const float*)d_in[4];
  const float* pp_w  = (const float*)d_in[5];
  const float* pp_b  = (const float*)d_in[6];
  const float* ln1_g = (const float*)d_in[7];
  const float* ln1_b = (const float*)d_in[8];
  const float* l1_w  = (const float*)d_in[9];
  const float* l1_b  = (const float*)d_in[10];
  const float* ln2_g = (const float*)d_in[11];
  const float* ln2_b = (const float*)d_in[12];
  const float* l2_w  = (const float*)d_in[13];
  const float* l2_b  = (const float*)d_in[14];
  const float* ln3_g = (const float*)d_in[15];
  const float* ln3_b = (const float*)d_in[16];
  const float* l3_w  = (const float*)d_in[17];
  const float* l3_b  = (const float*)d_in[18];
  float* out = (float*)d_out;

  // workspace layout (256B aligned)
  char* ws = (char*)d_ws;
  constexpr size_t OFF_RPB  = 0;                                   // 729*4 f32
  constexpr size_t OFF_BT   = 11776;                               // 4*13*224*16 f32
  constexpr size_t OFF_WQKV = OFF_BT  + 745472;                    // 384*128 bf16
  constexpr size_t OFF_WPRJ = OFF_WQKV + 384 * C_DIM * 2;          // 128*128 bf16
  constexpr size_t OFF_QK   = OFF_WPRJ + C_DIM * C_DIM * 2;        // MTOT*256 bf16
  constexpr size_t OFF_VT   = OFF_QK  + (size_t)MTOT * QK_LD * 2;  // B*128*224 bf16
  constexpr size_t OFF_AOUT = OFF_VT  + (size_t)BATCH * C_DIM * KT_PAD * 2;

  float*  rpb    = (float*)(ws + OFF_RPB);
  float*  Bt     = (float*)(ws + OFF_BT);
  __bf16* wqkvT  = (__bf16*)(ws + OFF_WQKV);
  __bf16* wprojT = (__bf16*)(ws + OFF_WPRJ);
  __bf16* qkbuf  = (__bf16*)(ws + OFF_QK);
  __bf16* vt     = (__bf16*)(ws + OFF_VT);
  __bf16* aoutb  = (__bf16*)(ws + OFF_AOUT);

  const int vt_u4 = (int)((size_t)BATCH * C_DIM * KT_PAD * 2 / 16);
  hlg_fill0<<<dim3((vt_u4 + 255) / 256), dim3(256), 0, stream>>>((uint4*)vt, vt_u4);

  hlg_pos_mlp<<<dim3(3), dim3(256), 0, stream>>>(
      pp_w, pp_b, ln1_g, ln1_b, l1_w, l1_b,
      ln2_g, ln2_b, l2_w, l2_b, ln3_g, ln3_b, l3_w, l3_b, rpb);

  hlg_bias_expand<<<dim3((H_HEADS * NKT * KT_PAD * 16 + 255) / 256), dim3(256), 0, stream>>>(rpb, Bt);

  hlg_cvt_w<<<dim3(256), dim3(256), 0, stream>>>(wq, wkv, wproj, wqkvT, wprojT);

  hlg_qkv_gemm<<<dim3(MTILES), dim3(256), 0, stream>>>(x, wqkvT, qkbuf, vt);

  hlg_attn<<<dim3(NKT, H_HEADS, BATCH), dim3(32), 0, stream>>>(qkbuf, vt, Bt, aoutb);

  hlg_proj_gemm<<<dim3(MTILES), dim3(256), 0, stream>>>(aoutb, wprojT, bproj, out);
}
